// MoDLayer_50689204027433
// MI455X (gfx1250) — matrix operand order repaired and checked by reading, not compiled, not run on hardware
//
#include <hip/hip_runtime.h>
#include <hip/hip_bf16.h>
#include <math.h>

typedef __attribute__((ext_vector_type(16))) _Float16 v16h;
typedef __attribute__((ext_vector_type(8)))  float    v8f;
typedef __attribute__((ext_vector_type(4)))  unsigned int u32x4;
typedef __attribute__((ext_vector_type(8)))  int          i32x8;
typedef __attribute__((ext_vector_type(4)))  int          i32x4;

#define DEV __device__ __forceinline__

constexpr int kB  = 2;
constexpr int kN  = 4096;
constexpr int kD  = 1024;
constexpr int kTK = 2048;   // top-k == capacity
constexpr int kH  = 16;
constexpr int kDH = 64;
constexpr int kS  = 2048;   // capacity sequence length
constexpr int k3D = 3 * kD;

DEV int laneid() { return (int)(threadIdx.x & 31); }

// A-tile 16x32 f16 per CDNA5 ISA layout:
// lanes 0-15: row=lane, halves = {K0..K7, K16..K23}
// lanes 16-31: row=lane-16, halves = {K8..K15, K24..K31}
DEV v16h load_A16x32(const _Float16* base, int ld, int lane) {
  int row = lane & 15;
  int kg  = (lane >> 4) & 1;
  const _Float16* p0 = base + (size_t)row * ld + kg * 8;
  v16h a;
  uint4* av = reinterpret_cast<uint4*>(&a);
  av[0] = *reinterpret_cast<const uint4*>(p0);
  av[1] = *reinterpret_cast<const uint4*>(p0 + 16);
  return a;
}

// B-tile 32x16 f16, loaded from a "BT" buffer where each output column's
// K-values are contiguous (bt[col*ldbt + k]).
DEV v16h load_BT32x16(const _Float16* bt, int ldbt, int lane) {
  int col = lane & 15;
  int kg  = (lane >> 4) & 1;
  const _Float16* p = bt + (size_t)col * ldbt + kg * 8;
  v16h b;
  uint4* bv = reinterpret_cast<uint4*>(&b);
  bv[0] = *reinterpret_cast<const uint4*>(p);
  bv[1] = *reinterpret_cast<const uint4*>(p + 16);
  return b;
}

DEV v8f wmma16(v16h a, v16h b, v8f c) {
  return __builtin_amdgcn_wmma_f32_16x16x32_f16(
      /*neg_a=*/false, a, /*neg_b=*/false, b,
      /*c_mod=*/(short)0, c, /*reuse_a=*/false, /*reuse_b=*/false);
}

// ---- Tensor Data Mover: DMA a 2D row-major f16 panel (rows x cols) into LDS.
// D# built per CDNA5 ISA ch.8: group0 = {count/type/lds_addr/global_addr},
// group1 = {data_size, tensor dims, tile dims, dim0 stride}. 2D -> groups 2/3 zero.
// 6-arg builtin (clang-23 / therock-10.0).
DEV void tdm_load_2d_f16(const _Float16* gsrc, void* lds_dst,
                         unsigned rows, unsigned cols, unsigned strideElems) {
  unsigned long long ga = (unsigned long long)(const void*)gsrc;
  unsigned lds = (unsigned)(unsigned long long)lds_dst;
  u32x4 g0;
  g0[0] = 1u;                                             // count=1 (valid user D#)
  g0[1] = lds;                                            // lds_addr (bytes)
  g0[2] = (unsigned)(ga & 0xffffffffu);                   // global_addr[31:0]
  g0[3] = (unsigned)((ga >> 32) & 0x01ffffffu) | (2u << 30); // global_addr[56:32], type=2
  i32x8 g1;
  g1[0] = (int)(1u << 16);                                // wg_mask=0, data_size=1 (2B)
  g1[1] = (int)((cols & 0xffffu) << 16);                  // tensor_dim0[15:0]
  g1[2] = (int)(((cols >> 16) & 0xffffu) | ((rows & 0xffffu) << 16)); // dim0 hi | tensor_dim1 lo
  g1[3] = (int)(((rows >> 16) & 0xffffu) | ((cols & 0xffffu) << 16)); // dim1 hi | tile_dim0
  g1[4] = (int)(rows & 0xffffu);                          // tile_dim1 | tile_dim2=0
  g1[5] = (int)strideElems;                               // tensor_dim0_stride[31:0]
  g1[6] = 0;                                              // stride hi | dim1_stride lo
  g1[7] = 0;                                              // dim1_stride hi
  i32x4 zz4 = {0, 0, 0, 0};
  i32x8 zz8 = {0, 0, 0, 0, 0, 0, 0, 0};
  __builtin_amdgcn_tensor_load_to_lds(g0, g1, zz4, zz4, zz8, 0);
}

// ---------------- router logits: one wave per token ----------------
__global__ void k_router(const float* __restrict__ x, const float* __restrict__ Wr,
                         float* __restrict__ logits) {
  int tok  = blockIdx.x;                 // over kB*kN
  int lane = threadIdx.x;
  const float* xr = x + (size_t)tok * kD;
  float s = 0.f;
  for (int i = lane; i < kD; i += 32) s += xr[i] * Wr[i];
#pragma unroll
  for (int m = 16; m >= 1; m >>= 1) s += __shfl_xor(s, m, 32);
  if (lane == 0) logits[tok] = s;
}

// ---------------- per-batch softmax stats (max, sumexp) ----------------
__global__ void k_stats(const float* __restrict__ logits, float* __restrict__ stats) {
  __shared__ float red[1024];
  int b = blockIdx.x, tid = threadIdx.x;
  const float* l = logits + (size_t)b * kN;
  float m = -INFINITY;
  for (int i = tid; i < kN; i += 1024) m = fmaxf(m, l[i]);
  red[tid] = m; __syncthreads();
  for (int s = 512; s > 0; s >>= 1) {
    if (tid < s) red[tid] = fmaxf(red[tid], red[tid + s]);
    __syncthreads();
  }
  float mx = red[0]; __syncthreads();
  float sum = 0.f;
  for (int i = tid; i < kN; i += 1024) sum += __expf(l[i] - mx);
  red[tid] = sum; __syncthreads();
  for (int s = 512; s > 0; s >>= 1) {
    if (tid < s) red[tid] += red[tid + s];
    __syncthreads();
  }
  if (tid == 0) { stats[b * 2] = mx; stats[b * 2 + 1] = red[0]; }
}

// -------- top-k via full bitonic sort in LDS, then ascending index sort -----
__global__ void k_topk(const float* __restrict__ logits, const float* __restrict__ stats,
                       int* __restrict__ sel, float* __restrict__ wgt) {
  __shared__ float skey[kN];
  __shared__ int   sidx[kN];
  int b = blockIdx.x, tid = threadIdx.x, nt = blockDim.x;
  const float* l = logits + (size_t)b * kN;
  for (int i = tid; i < kN; i += nt) { skey[i] = l[i]; sidx[i] = i; }
  __syncthreads();
  // sort descending by key (tie -> smaller index first), full sort of 4096
  for (int k = 2; k <= kN; k <<= 1) {
    for (int j = k >> 1; j > 0; j >>= 1) {
      for (int t = tid; t < kN; t += nt) {
        int p = t ^ j;
        if (p > t) {
          bool up = ((t & k) == 0);
          float ka = skey[t], kb = skey[p];
          int   ia = sidx[t], ib = sidx[p];
          bool aFirst = (ka > kb) || (ka == kb && ia < ib);
          bool sw = up ? !aFirst : aFirst;
          if (sw) { skey[t] = kb; sidx[t] = ib; skey[p] = ka; sidx[p] = ia; }
        }
      }
      __syncthreads();
    }
  }
  // ascending sort of the selected 2048 indices
  for (int k = 2; k <= kTK; k <<= 1) {
    for (int j = k >> 1; j > 0; j >>= 1) {
      for (int t = tid; t < kTK; t += nt) {
        int p = t ^ j;
        if (p > t) {
          bool up = ((t & k) == 0);
          int ia = sidx[t], ib = sidx[p];
          bool sw = up ? (ia > ib) : (ia < ib);
          if (sw) { sidx[t] = ib; sidx[p] = ia; }
        }
      }
      __syncthreads();
    }
  }
  float mx = stats[b * 2], sm = stats[b * 2 + 1];
  for (int t = tid; t < kTK; t += nt) {
    int tok = sidx[t];
    sel[(size_t)b * kTK + t] = tok;
    wgt[(size_t)b * kTK + t] = __expf(l[tok] - mx) / sm;
  }
}

// ---------------- weight transpose + f32->f16 convert ----------------
__global__ void k_transpose(const float* __restrict__ W, _Float16* __restrict__ WT,
                            int K, int Ncol) {
  size_t i = blockIdx.x * (size_t)blockDim.x + threadIdx.x;
  if (i >= (size_t)K * Ncol) return;
  int n = (int)(i % Ncol);
  int k = (int)(i / Ncol);
  WT[(size_t)n * K + k] = (_Float16)W[i];
}

// ---------------- out = x (residual init) ----------------
__global__ void k_copy(const float* __restrict__ x, float* __restrict__ out, size_t n4) {
  size_t i = blockIdx.x * (size_t)blockDim.x + threadIdx.x;
  if (i < n4) reinterpret_cast<float4*>(out)[i] = reinterpret_cast<const float4*>(x)[i];
}

// ---------------- gather selected tokens, f32 -> f16 ----------------
__global__ void k_gather(const float* __restrict__ x, const int* __restrict__ sel,
                         _Float16* __restrict__ xg) {
  size_t i = blockIdx.x * (size_t)blockDim.x + threadIdx.x;  // 4 elems/thread
  size_t total = (size_t)kB * kTK * (kD / 4);
  if (i >= total) return;
  int d4 = (int)(i % (kD / 4));
  int s  = (int)((i / (kD / 4)) % kTK);
  int b  = (int)(i / ((size_t)(kD / 4) * kTK));
  int tok = sel[(size_t)b * kTK + s];
  float4 v = reinterpret_cast<const float4*>(x + ((size_t)b * kN + tok) * kD)[d4];
  _Float16* o = xg + ((size_t)b * kTK + s) * kD + d4 * 4;
  o[0] = (_Float16)v.x; o[1] = (_Float16)v.y; o[2] = (_Float16)v.z; o[3] = (_Float16)v.w;
}

// Ping-pong WMMA GEMM core: A panel in LDS (TDM-staged), B streamed from
// global, unroll-by-2 so no register-rotation copies are needed.
DEV void gemm_core(const _Float16* lA, const _Float16* btBase, int lane, v8f acc[4]) {
  v16h aA, aB, bA[4], bB[4];
#pragma unroll
  for (int c = 0; c < 4; ++c) bA[c] = load_BT32x16(btBase + (size_t)c * 16 * kD, kD, lane);
  __builtin_amdgcn_s_wait_tensorcnt(0);          // A panel resident in LDS
  aA = load_A16x32(lA, kD, lane);
  for (int k0 = 0; k0 < kD; k0 += 64) {
    // prefetch step k0+32 into set B (always valid: kD multiple of 64)
#pragma unroll
    for (int c = 0; c < 4; ++c)
      bB[c] = load_BT32x16(btBase + (size_t)c * 16 * kD + k0 + 32, kD, lane);
    aB = load_A16x32(lA + k0 + 32, kD, lane);
#pragma unroll
    for (int c = 0; c < 4; ++c) acc[c] = wmma16(aA, bA[c], acc[c]);
    // prefetch step k0+64 into set A
    if (k0 + 64 < kD) {
#pragma unroll
      for (int c = 0; c < 4; ++c)
        bA[c] = load_BT32x16(btBase + (size_t)c * 16 * kD + k0 + 64, kD, lane);
      aA = load_A16x32(lA + k0 + 64, kD, lane);
    }
#pragma unroll
    for (int c = 0; c < 4; ++c) acc[c] = wmma16(aB, bB[c], acc[c]);
  }
}

// ---- QKV GEMM: TDM-staged A panel in LDS, ping-pong B, WMMA core ----
__global__ void k_gemm_qkv(const _Float16* __restrict__ xg, const _Float16* __restrict__ WqkvT,
                           _Float16* __restrict__ qh, _Float16* __restrict__ kh,
                           _Float16* __restrict__ vth) {
  __shared__ _Float16 lA[16 * kD];               // 32 KB A panel
  const int nCG = k3D / 64;                      // 48 groups of 4 column tiles
  int cg = blockIdx.x % nCG;
  int rt = (blockIdx.x / nCG) % (kS / 16);
  int b  = blockIdx.x / (nCG * (kS / 16));
  int lane = laneid();
  const _Float16* A = xg + ((size_t)b * kS + rt * 16) * kD;
  tdm_load_2d_f16(A, lA, 16, kD, kD);            // async DMA panel -> LDS
  const _Float16* btBase = WqkvT + (size_t)(cg * 4) * 16 * kD;
  v8f z = {0.f, 0.f, 0.f, 0.f, 0.f, 0.f, 0.f, 0.f};
  v8f acc[4];
#pragma unroll
  for (int c = 0; c < 4; ++c) acc[c] = z;
  gemm_core(lA, btBase, lane, acc);
  int colN = lane & 15;
  int rowB = (lane >> 4) * 8;
#pragma unroll
  for (int c = 0; c < 4; ++c) {
    int noutb = (cg * 4 + c) * 16;
    int part  = noutb >> 10;                 // 0=q 1=k 2=v
    int dcolb = noutb & (kD - 1);
    int h     = dcolb >> 6;
    int dcol  = (dcolb & 63) + colN;
#pragma unroll
    for (int j = 0; j < 8; ++j) {
      int s = rt * 16 + rowB + j;
      float v = acc[c][j];
      if (part == 0)        // q pre-scaled by 1/sqrt(Dh)
        qh[(((size_t)b * kH + h) * kS + s) * kDH + dcol] = (_Float16)(v * 0.125f);
      else if (part == 1)
        kh[(((size_t)b * kH + h) * kS + s) * kDH + dcol] = (_Float16)v;
      else                  // v stored transposed [Dh][S] for contiguous PV B-tiles
        vth[(((size_t)b * kH + h) * kDH + dcol) * kS + s] = (_Float16)v;
    }
  }
}

// ---------------- flash attention: one wave = 16 query rows x full Dh --------
__global__ void k_flash(const _Float16* __restrict__ qh, const _Float16* __restrict__ kh,
                        const _Float16* __restrict__ vth, _Float16* __restrict__ attn) {
  __shared__ _Float16 pl[16 * 32];               // probs staging, C->A relayout
  int qt = blockIdx.x % (kS / 16);
  int bh = blockIdx.x / (kS / 16);
  int h = bh % kH, b = bh / kH;
  int lane = laneid();
  const _Float16* q  = qh  + (((size_t)b * kH + h) * kS + qt * 16) * kDH;
  const _Float16* kk = kh  + ((size_t)b * kH + h) * (size_t)kS * kDH;
  const _Float16* vt = vth + ((size_t)b * kH + h) * (size_t)kDH * kS;
  v16h a0 = load_A16x32(q, kDH, lane);           // d 0..31
  v16h a1 = load_A16x32(q + 32, kDH, lane);      // d 32..63
  v8f z = {0.f, 0.f, 0.f, 0.f, 0.f, 0.f, 0.f, 0.f};
  v8f o0 = z, o1 = z, o2 = z, o3 = z;
  float rmax[8], rsum[8];
#pragma unroll
  for (int j = 0; j < 8; ++j) { rmax[j] = -INFINITY; rsum[j] = 0.f; }
  int colN = lane & 15;
  int rowB = (lane >> 4) * 8;
  for (int n0 = 0; n0 < kS; n0 += 32) {
    // V B-tiles issued first: in flight during score WMMAs + online softmax
    v16h vb0 = load_BT32x16(vt + (size_t)0  * kS + n0, kS, lane);
    v16h vb1 = load_BT32x16(vt + (size_t)16 * kS + n0, kS, lane);
    v16h vb2 = load_BT32x16(vt + (size_t)32 * kS + n0, kS, lane);
    v16h vb3 = load_BT32x16(vt + (size_t)48 * kS + n0, kS, lane);
    v8f c0 = z, c1 = z;                           // scores, 16x32 over key chunk
    c0 = wmma16(a0, load_BT32x16(kk + (size_t)n0 * kDH,            kDH, lane), c0);
    c0 = wmma16(a1, load_BT32x16(kk + (size_t)n0 * kDH + 32,       kDH, lane), c0);
    c1 = wmma16(a0, load_BT32x16(kk + (size_t)(n0 + 16) * kDH,     kDH, lane), c1);
    c1 = wmma16(a1, load_BT32x16(kk + (size_t)(n0 + 16) * kDH + 32, kDH, lane), c1);
    float alpha[8];
#pragma unroll
    for (int j = 0; j < 8; ++j) {
      float lm = fmaxf(c0[j], c1[j]);
#pragma unroll
      for (int m = 8; m >= 1; m >>= 1) lm = fmaxf(lm, __shfl_xor(lm, m, 32));
      float nm = fmaxf(rmax[j], lm);
      float p0 = __expf(c0[j] - nm);
      float p1 = __expf(c1[j] - nm);
      pl[(rowB + j) * 32 + colN]      = (_Float16)p0;
      pl[(rowB + j) * 32 + 16 + colN] = (_Float16)p1;
      float ts = p0 + p1;
#pragma unroll
      for (int m = 8; m >= 1; m >>= 1) ts += __shfl_xor(ts, m, 32);
      alpha[j] = __expf(rmax[j] - nm);
      rsum[j]  = rsum[j] * alpha[j] + ts;
      rmax[j]  = nm;
    }
#pragma unroll
    for (int j = 0; j < 8; ++j) {
      o0[j] *= alpha[j]; o1[j] *= alpha[j]; o2[j] *= alpha[j]; o3[j] *= alpha[j];
    }
    __syncthreads();
    v16h ap = load_A16x32(pl, 32, lane);          // probs as A-tile via LDS
    o0 = wmma16(ap, vb0, o0);
    o1 = wmma16(ap, vb1, o1);
    o2 = wmma16(ap, vb2, o2);
    o3 = wmma16(ap, vb3, o3);
    __syncthreads();
  }
#pragma unroll
  for (int j = 0; j < 8; ++j) {
    float inv = 1.f / rsum[j];
    int s = qt * 16 + rowB + j;
    _Float16* dst = attn + ((size_t)b * kS + s) * kD + h * kDH;
    dst[colN]      = (_Float16)(o0[j] * inv);
    dst[16 + colN] = (_Float16)(o1[j] * inv);
    dst[32 + colN] = (_Float16)(o2[j] * inv);
    dst[48 + colN] = (_Float16)(o3[j] * inv);
  }
}

// ------ O projection GEMM + weighted scatter-add into the residual stream ----
__global__ void k_gemm_out(const _Float16* __restrict__ attn, const _Float16* __restrict__ WoT,
                           const int* __restrict__ sel, const float* __restrict__ wgt,
                           float* __restrict__ out) {
  __shared__ _Float16 lA[16 * kD];                // 32 KB A panel
  const int nCG = kD / 64;                        // 16 groups of 4 column tiles
  int cg = blockIdx.x % nCG;
  int rt = (blockIdx.x / nCG) % (kS / 16);
  int b  = blockIdx.x / (nCG * (kS / 16));
  int lane = laneid();
  const _Float16* A = attn + ((size_t)b * kS + rt * 16) * kD;
  tdm_load_2d_f16(A, lA, 16, kD, kD);
  const _Float16* btBase = WoT + (size_t)(cg * 4) * 16 * kD;
  v8f z = {0.f, 0.f, 0.f, 0.f, 0.f, 0.f, 0.f, 0.f};
  v8f acc[4];
#pragma unroll
  for (int c = 0; c < 4; ++c) acc[c] = z;
  gemm_core(lA, btBase, lane, acc);
  int colN = lane & 15;
  int rowB = (lane >> 4) * 8;
  int   tok[8];
  float wv[8];
#pragma unroll
  for (int j = 0; j < 8; ++j) {
    int s = rt * 16 + rowB + j;
    tok[j] = sel[(size_t)b * kTK + s];
    wv[j]  = wgt[(size_t)b * kTK + s];
  }
#pragma unroll
  for (int c = 0; c < 4; ++c) {
    int dout = (cg * 4 + c) * 16 + colN;
#pragma unroll
    for (int j = 0; j < 8; ++j) {
      // each (b, sel-row, dout) touched by exactly one lane -> plain RMW is safe
      float* dst = out + ((size_t)b * kN + tok[j]) * kD + dout;
      *dst += wv[j] * acc[c][j];
    }
  }
}

extern "C" void kernel_launch(void* const* d_in, const int* in_sizes, int n_in,
                              void* d_out, int out_size, void* d_ws, size_t ws_size,
                              hipStream_t stream) {
  const float* x    = (const float*)d_in[0];
  const float* Wr   = (const float*)d_in[1];
  const float* Wqkv = (const float*)d_in[2];
  const float* Wo   = (const float*)d_in[3];
  float* out = (float*)d_out;
  (void)in_sizes; (void)n_in; (void)out_size; (void)ws_size;

  char* ws = (char*)d_ws;
  size_t off = 0;
  auto take = [&](size_t bytes) {
    char* p = ws + off;
    off = (off + bytes + 255) & ~(size_t)255;
    return p;
  };
  float*    logits = (float*)take((size_t)kB * kN * 4);
  float*    stats  = (float*)take((size_t)kB * 2 * 4);
  int*      sel    = (int*)take((size_t)kB * kTK * 4);
  float*    wgt    = (float*)take((size_t)kB * kTK * 4);
  _Float16* WqkvT  = (_Float16*)take((size_t)k3D * kD * 2);
  _Float16* WoT    = (_Float16*)take((size_t)kD * kD * 2);
  _Float16* xg     = (_Float16*)take((size_t)kB * kTK * kD * 2);
  _Float16* qh     = (_Float16*)take((size_t)kB * kH * kS * kDH * 2);
  _Float16* kh     = (_Float16*)take((size_t)kB * kH * kS * kDH * 2);
  _Float16* vth    = (_Float16*)take((size_t)kB * kH * kDH * kS * 2);
  _Float16* attn   = (_Float16*)take((size_t)kB * kS * kD * 2);

  hipLaunchKernelGGL(k_transpose, dim3((kD * k3D + 255) / 256), dim3(256), 0, stream,
                     Wqkv, WqkvT, kD, k3D);
  hipLaunchKernelGGL(k_transpose, dim3((kD * kD + 255) / 256), dim3(256), 0, stream,
                     Wo, WoT, kD, kD);
  hipLaunchKernelGGL(k_copy, dim3((kB * kN * kD / 4 + 255) / 256), dim3(256), 0, stream,
                     x, out, (size_t)kB * kN * kD / 4);
  hipLaunchKernelGGL(k_router, dim3(kB * kN), dim3(32), 0, stream, x, Wr, logits);
  hipLaunchKernelGGL(k_stats, dim3(kB), dim3(1024), 0, stream, logits, stats);
  hipLaunchKernelGGL(k_topk, dim3(kB), dim3(1024), 0, stream, logits, stats, sel, wgt);
  hipLaunchKernelGGL(k_gather, dim3((kB * kTK * (kD / 4) + 255) / 256), dim3(256), 0, stream,
                     x, sel, xg);
  hipLaunchKernelGGL(k_gemm_qkv, dim3(kB * (kS / 16) * (k3D / 64)), dim3(32), 0, stream,
                     xg, WqkvT, qh, kh, vth);
  hipLaunchKernelGGL(k_flash, dim3(kB * kH * (kS / 16)), dim3(32), 0, stream,
                     qh, kh, vth, attn);
  hipLaunchKernelGGL(k_gemm_out, dim3(kB * (kS / 16) * (kD / 64)), dim3(32), 0, stream,
                     attn, WoT, sel, wgt, out);
}
